// FractalAttention_13932873908335
// MI455X (gfx1250) — compile-verified, matrix operand order checked by reading, not run on hardware
//
#include <hip/hip_runtime.h>
#include <hip/hip_bf16.h>

// ---------------------------------------------------------------------------
// FractalAttention on MI455X (gfx1250).
// Heavy math: v_wmma_f32_16x16x32_f16 (wave32 WMMA), f32 accumulate.
// Data movement: GLOBAL_LOAD_ASYNC_TO_LDS_B128 double-buffered tiles
// (ASYNCcnt), operands fed from LDS via ds_load_b128.
// ---------------------------------------------------------------------------

typedef __attribute__((ext_vector_type(16))) _Float16 v16h;
typedef __attribute__((ext_vector_type(8)))  float    v8f;
typedef __attribute__((ext_vector_type(4)))  int      v4i;

union Frag {
    v16h  h;
    uint4 q[2];
};

__device__ __forceinline__ v8f wmma16(v16h a, v16h b, v8f c) {
    // (neg_a, A, neg_b, B, c_mod, C, reuse_a, reuse_b)
    return __builtin_amdgcn_wmma_f32_16x16x32_f16(false, a, false, b, (short)0, c,
                                                  false, false);
}

// ---- async global -> LDS copy (16 bytes / thread / issue) ------------------
#if __has_builtin(__builtin_amdgcn_global_load_async_to_lds_b128)
typedef __attribute__((address_space(1))) v4i* gv4p;   // prints as "__device__ v4i*"
typedef __attribute__((address_space(3))) v4i* lv4p;   // prints as "__shared__ v4i*"
__device__ __forceinline__ void async_cp16(const _Float16* g, _Float16* l) {
    __builtin_amdgcn_global_load_async_to_lds_b128(
        (gv4p)(v4i*)(void*)const_cast<_Float16*>(g),
        (lv4p)(v4i*)(void*)l, 0, 0);
}
#else
__device__ __forceinline__ void async_cp16(const _Float16* g, _Float16* l) {
    unsigned loff = (unsigned)(unsigned long long)l;  // LDS byte offset (low 32 bits)
    asm volatile("global_load_async_to_lds_b128 %0, %1, off"
                 :: "v"(loff), "v"(g) : "memory");
}
#endif

#if __has_builtin(__builtin_amdgcn_s_wait_asynccnt)
#define WAIT_ASYNC(n) __builtin_amdgcn_s_wait_asynccnt(n)
#else
#define WAIT_ASYNC(n) asm volatile("s_wait_asynccnt %0" :: "i"(n) : "memory")
#endif

// Problem constants
#define BB  4
#define LL  4096
#define DD  1024
#define HH  16
#define SS  4
#define DH  64
#define MM  (BB * LL)            // 16384 rows for flattened (B,L)

// GEMM tiling
#define MBLK 128                 // block tile M
#define NBLK 64                  // block tile N
#define KCH  32                  // K chunk (one WMMA K per chunk)
#define LPAD 40                  // padded LDS row (halves): 32 data + 8 pad (80B rows)

// ---------------------------------------------------------------------------
// f32 -> f16 conversion (grid-stride)
// ---------------------------------------------------------------------------
__global__ void k_cvt_f16(const float* __restrict__ src, _Float16* __restrict__ dst,
                          long long n) {
    long long i = (long long)blockIdx.x * blockDim.x + threadIdx.x;
    long long stride = (long long)gridDim.x * blockDim.x;
    for (; i < n; i += stride) dst[i] = (_Float16)src[i];
}

// ---------------------------------------------------------------------------
// GEMM: C[z] = A (MxK, f16, row-major) * W[z] (NxK, f16, row-major)^T + bias[z]
// 256 threads = 8 waves. Block tile 128x64; each wave owns a 32x32 output
// (2x2 grid of 16x16 WMMA accumulators). Per K-chunk of 32:
//   - async-copy A(128x32) and B(64x32) tiles to LDS (double buffered)
//   - each wave: 8x ds_load_b128 fragments + 4x v_wmma_f32_16x16x32_f16
// grid = (N/64, M/128, Z)
// ---------------------------------------------------------------------------
__global__ __launch_bounds__(256) void k_gemm(const _Float16* __restrict__ A,
                                              const _Float16* __restrict__ W,
                                              const float* __restrict__ bias,
                                              void* __restrict__ C,
                                              int M, int N, int K,
                                              long long wz, long long bz, long long cz,
                                              int storeF32) {
    __shared__ _Float16 Al[2][MBLK][LPAD];   // 2 x 128 x 40 halves = 20 KB
    __shared__ _Float16 Bl[2][NBLK][LPAD];   // 2 x  64 x 40 halves = 10 KB

    const int tid  = threadIdx.x;
    const int lane = tid & 31;
    const int wave = tid >> 5;
    const int z    = blockIdx.z;

    const _Float16* Az = A;
    const _Float16* Wz = W + (long long)z * wz;
    const float*    Bz = bias ? (bias + (long long)z * bz) : nullptr;

    const int gn0 = blockIdx.x * NBLK;
    const int gm0 = blockIdx.y * MBLK;

    // async-copy assignment: 16B chunks; A tile has 512 chunks, B tile 256.
    const int crow = tid >> 2;             // 0..63
    const int ccol = (tid & 3) * 8;        // half offset within 32-wide chunk row
    const _Float16* gA0 = Az + (size_t)(gm0 + crow) * K + ccol;
    const _Float16* gA1 = Az + (size_t)(gm0 + crow + 64) * K + ccol;
    const _Float16* gB0 = Wz + (size_t)(gn0 + crow) * K + ccol;

    // wave -> 32x32 subtile: 4 M-waves x 2 N-waves
    const int wm = (wave & 3) * 32;
    const int wn = (wave >> 2) * 32;
    const int row16 = lane & 15;
    const int koff  = (lane >> 4) * 8;     // 16-bit A layout: hi half-wave holds K+8

    v8f acc00 = {}, acc01 = {}, acc10 = {}, acc11 = {};

    const int nch = K / KCH;

    // prologue: chunk 0 -> buffer 0
    async_cp16(gA0, &Al[0][crow][ccol]);
    async_cp16(gA1, &Al[0][crow + 64][ccol]);
    async_cp16(gB0, &Bl[0][crow][ccol]);

    for (int ck = 0; ck < nch; ++ck) {
        const int buf = ck & 1;
        if (ck + 1 < nch) {
            const int kb = (ck + 1) * KCH;
            async_cp16(gA0 + kb, &Al[buf ^ 1][crow][ccol]);
            async_cp16(gA1 + kb, &Al[buf ^ 1][crow + 64][ccol]);
            async_cp16(gB0 + kb, &Bl[buf ^ 1][crow][ccol]);
            WAIT_ASYNC(3);      // FIFO: chunk ck's 3 copies have landed
        } else {
            WAIT_ASYNC(0);
        }
        __syncthreads();

        Frag a0, a1, b0, b1;
        {
            const _Float16* p = &Al[buf][wm + row16][koff];
            a0.q[0] = *(const uint4*)p;        a0.q[1] = *(const uint4*)(p + 16);
            p = &Al[buf][wm + 16 + row16][koff];
            a1.q[0] = *(const uint4*)p;        a1.q[1] = *(const uint4*)(p + 16);
            p = &Bl[buf][wn + row16][koff];
            b0.q[0] = *(const uint4*)p;        b0.q[1] = *(const uint4*)(p + 16);
            p = &Bl[buf][wn + 16 + row16][koff];
            b1.q[0] = *(const uint4*)p;        b1.q[1] = *(const uint4*)(p + 16);
        }
        acc00 = wmma16(a0.h, b0.h, acc00);
        acc01 = wmma16(a0.h, b1.h, acc01);
        acc10 = wmma16(a1.h, b0.h, acc10);
        acc11 = wmma16(a1.h, b1.h, acc11);

        __syncthreads();   // protect buf from being overwritten next iteration
    }

    // store: C layout VGPR r -> M = r + 8*(lane>=16), N = lane&15
    const int chi = lane >> 4;
    const int col = lane & 15;
    v8f accs[2][2] = {{acc00, acc01}, {acc10, acc11}};
#pragma unroll
    for (int i = 0; i < 2; ++i) {
#pragma unroll
        for (int j = 0; j < 2; ++j) {
            const int n = gn0 + wn + j * 16 + col;
            const float bv = Bz ? Bz[n] : 0.0f;
#pragma unroll
            for (int r = 0; r < 8; ++r) {
                const int m = gm0 + wm + i * 16 + r + 8 * chi;
                const float v = accs[i][j][r] + bv;
                if (storeF32)
                    ((float*)C)[(size_t)z * cz + (size_t)m * N + n] = v;
                else
                    ((_Float16*)C)[(size_t)z * cz + (size_t)m * N + n] = (_Float16)v;
            }
        }
    }
}

// ---------------------------------------------------------------------------
// Per-position head-mixing attention. One wave per (s,b,l) position:
//   scores(16x16) = sq(16x64) @ sk(16x64)^T * 1/8   -> 2 chained WMMAs (K=64)
//   attn = softmax_rows(scores)                     -> shfl_xor reductions
//   out(16x64)   = attn(16x16) @ sv(16x64)          -> 4 WMMAs (K padded to 32)
// ---------------------------------------------------------------------------
__global__ __launch_bounds__(128) void k_attn(const _Float16* __restrict__ sq,
                                              const _Float16* __restrict__ sk,
                                              const _Float16* __restrict__ sv,
                                              _Float16* __restrict__ outp,
                                              int P) {
    __shared__ _Float16 attn_s[4][16 * 16];

    const int lane = threadIdx.x & 31;
    const int wave = threadIdx.x >> 5;
    int p = blockIdx.x * 4 + wave;
    if (p >= P) p = P - 1;   // keep all lanes live (EXEC must be all-1s for WMMA)

    const size_t base = (size_t)p * (HH * DH);
    const int row  = lane & 15;
    const int chi  = lane >> 4;
    const int koff = chi * 8;

    const _Float16* qr = sq + base + (size_t)row * DH + koff;
    const _Float16* kr = sk + base + (size_t)row * DH + koff;
    Frag a0, a1, b0, b1;
    a0.q[0] = *(const uint4*)(qr);      a0.q[1] = *(const uint4*)(qr + 16);
    a1.q[0] = *(const uint4*)(qr + 32); a1.q[1] = *(const uint4*)(qr + 48);
    b0.q[0] = *(const uint4*)(kr);      b0.q[1] = *(const uint4*)(kr + 16);
    b1.q[0] = *(const uint4*)(kr + 32); b1.q[1] = *(const uint4*)(kr + 48);

    v8f c = {};
    c = wmma16(a0.h, b0.h, c);
    c = wmma16(a1.h, b1.h, c);

    const float scale = 0.125f;   // 1/sqrt(64)
    float prob[8];
#pragma unroll
    for (int r = 0; r < 8; ++r) {
        float s  = c[r] * scale;
        float mx = s;
#pragma unroll
        for (int off = 1; off < 16; off <<= 1) mx = fmaxf(mx, __shfl_xor(mx, off, 16));
        float e   = __expf(s - mx);
        float sum = e;
#pragma unroll
        for (int off = 1; off < 16; off <<= 1) sum += __shfl_xor(sum, off, 16);
        prob[r] = e / sum;
    }

    // re-layout attn (C layout -> A layout) through LDS
#pragma unroll
    for (int r = 0; r < 8; ++r)
        attn_s[wave][(r + 8 * chi) * 16 + (lane & 15)] = (_Float16)prob[r];
    __syncthreads();

    v16h av;
#pragma unroll
    for (int j = 0; j < 8; ++j) av[j] = attn_s[wave][row * 16 + koff + j];
#pragma unroll
    for (int j = 8; j < 16; ++j) av[j] = (_Float16)0.0f;   // pad K 16 -> 32

#pragma unroll
    for (int t = 0; t < 4; ++t) {
        const int n = t * 16 + (lane & 15);
        v16h bvv;
#pragma unroll
        for (int j = 0; j < 8; ++j)           // K = koff + j (g index)
            bvv[j] = sv[base + (size_t)(koff + j) * DH + n];
#pragma unroll
        for (int j = 8; j < 16; ++j) bvv[j] = (_Float16)0.0f;

        v8f o = {};
        o = wmma16(av, bvv, o);
#pragma unroll
        for (int r = 0; r < 8; ++r)
            outp[base + (size_t)(r + 8 * chi) * DH + n] = (_Float16)o[r];
    }
}

// ---------------------------------------------------------------------------
__global__ void k_scale_softmax(const float* __restrict__ logits, float* __restrict__ w) {
    if (threadIdx.x == 0 && blockIdx.x == 0) {
        float mx = logits[0];
        for (int s = 1; s < SS; ++s) mx = fmaxf(mx, logits[s]);
        float e[SS], sum = 0.0f;
        for (int s = 0; s < SS; ++s) { e[s] = __expf(logits[s] - mx); sum += e[s]; }
        for (int s = 0; s < SS; ++s) w[s] = e[s] / sum;
    }
}

__global__ void k_combine(const _Float16* __restrict__ outp, const float* __restrict__ w,
                          _Float16* __restrict__ comb, long long n, long long sstride) {
    long long i = (long long)blockIdx.x * blockDim.x + threadIdx.x;
    long long stride = (long long)gridDim.x * blockDim.x;
    for (; i < n; i += stride) {
        float acc = 0.0f;
#pragma unroll
        for (int s = 0; s < SS; ++s)
            acc += w[s] * (float)outp[(long long)s * sstride + i];
        comb[i] = (_Float16)acc;
    }
}

// ---------------------------------------------------------------------------
// Host-side orchestration
// ---------------------------------------------------------------------------
extern "C" void kernel_launch(void* const* d_in, const int* in_sizes, int n_in,
                              void* d_out, int out_size, void* d_ws, size_t ws_size,
                              hipStream_t stream) {
    (void)in_sizes; (void)n_in; (void)out_size; (void)ws_size;

    const float* x  = (const float*)d_in[0];
    const float* sl = (const float*)d_in[1];
    const float* wq = (const float*)d_in[2];
    const float* bq = (const float*)d_in[3];
    const float* wk = (const float*)d_in[4];
    const float* bk = (const float*)d_in[5];
    const float* wv = (const float*)d_in[6];
    const float* bv = (const float*)d_in[7];
    const float* wo = (const float*)d_in[8];
    const float* bo = (const float*)d_in[9];
    const float* ws = (const float*)d_in[10];
    const float* bs = (const float*)d_in[11];

    const long long MD  = (long long)MM * DD;       // 16,777,216 elems
    const long long D2  = (long long)DD * DD;       // 1,048,576 elems
    const long long SMD = (long long)SS * MD;       // 67,108,864 elems

    char* base = (char*)d_ws;
    size_t off = 0;
    auto alloc = [&](size_t bytes) { char* p = base + off; off = (off + bytes + 255) & ~(size_t)255; return p; };
    _Float16* xb   = (_Float16*)alloc(2 * MD);
    _Float16* wqb  = (_Float16*)alloc(2 * D2);
    _Float16* wkb  = (_Float16*)alloc(2 * D2);
    _Float16* wvb  = (_Float16*)alloc(2 * D2);
    _Float16* wob  = (_Float16*)alloc(2 * D2);
    _Float16* wsb  = (_Float16*)alloc(2 * SS * D2);
    _Float16* qf   = (_Float16*)alloc(2 * MD);
    _Float16* kf   = (_Float16*)alloc(2 * MD);
    _Float16* vf   = (_Float16*)alloc(2 * MD);
    _Float16* sq   = (_Float16*)alloc(2 * SMD);
    _Float16* sk   = (_Float16*)alloc(2 * SMD);
    _Float16* sv   = (_Float16*)alloc(2 * SMD);
    _Float16* outb = (_Float16*)alloc(2 * SMD);
    _Float16* comb = (_Float16*)alloc(2 * MD);
    float*    wsm  = (float*)alloc(256);

    // 1) convert inputs/weights to f16
    k_cvt_f16<<<4096, 256, 0, stream>>>(x,  xb,  MD);
    k_cvt_f16<<<1024, 256, 0, stream>>>(wq, wqb, D2);
    k_cvt_f16<<<1024, 256, 0, stream>>>(wk, wkb, D2);
    k_cvt_f16<<<1024, 256, 0, stream>>>(wv, wvb, D2);
    k_cvt_f16<<<1024, 256, 0, stream>>>(wo, wob, D2);
    k_cvt_f16<<<2048, 256, 0, stream>>>(ws, wsb, SS * D2);

    // 2) qkv projections
    dim3 gA(DD / NBLK, MM / MBLK, 1);
    k_gemm<<<gA, 256, 0, stream>>>(xb, wqb, bq, qf, MM, DD, DD, 0, 0, 0, 0);
    k_gemm<<<gA, 256, 0, stream>>>(xb, wkb, bk, kf, MM, DD, DD, 0, 0, 0, 0);
    k_gemm<<<gA, 256, 0, stream>>>(xb, wvb, bv, vf, MM, DD, DD, 0, 0, 0, 0);

    // 3) per-scale projections, batched over z = scale index
    dim3 gS(DD / NBLK, MM / MBLK, SS);
    k_gemm<<<gS, 256, 0, stream>>>(qf, wsb, bs, sq, MM, DD, DD, D2, DD, MD, 0);
    k_gemm<<<gS, 256, 0, stream>>>(kf, wsb, bs, sk, MM, DD, DD, D2, DD, MD, 0);
    k_gemm<<<gS, 256, 0, stream>>>(vf, wsb, bs, sv, MM, DD, DD, D2, DD, MD, 0);

    // 4) per-position 16x16 head-mixing attention (65536 positions, 4/block)
    const int P = SS * MM;
    k_attn<<<P / 4, 128, 0, stream>>>(sq, sk, sv, outb, P);

    // 5) scale softmax + weighted combine over scales
    k_scale_softmax<<<1, 32, 0, stream>>>(sl, wsm);
    k_combine<<<4096, 256, 0, stream>>>(outb, wsm, comb, MD, MD);

    // 6) output projection -> f32 result
    k_gemm<<<gA, 256, 0, stream>>>(comb, wob, bo, (float*)d_out, MM, DD, DD, 0, 0, 0, 1);
}
